// RuiCNRelaxedLiftingConvolution_25220047962411
// MI455X (gfx1250) — matrix operand order repaired and checked by reading, not compiled, not run on hardware
//
#include <hip/hip_runtime.h>
#include <hip/hip_bf16.h>
#include <math.h>
#include <stdint.h>

typedef __attribute__((ext_vector_type(16))) _Float16 v16h;
typedef __attribute__((ext_vector_type(8)))  float    v8f;
typedef __attribute__((ext_vector_type(4)))  unsigned int v4u;
typedef __attribute__((ext_vector_type(8)))  int      v8i;
typedef __attribute__((ext_vector_type(4)))  int      v4i;

#define B_      16
#define IC_     32
#define HW_     128
#define NF_     8
#define OC_     64
#define K_      5
#define GROUP_  8
#define CH_     (OC_ * GROUP_)   // 512
#define TAPS    (K_ * K_)        // 25
#define PAD_HW  (HW_ + 4)        // 132 (2-halo each side)

// Patch staged per block by the TDM: 5 rows x 68 pixels x 32 ch (f16)
#define PATCH_PIX   68
#define PATCH_ROW_E (PATCH_PIX * IC_)        // 2176 elements, contiguous in memory
#define PATCH_E     (K_ * PATCH_ROW_E)       // 10880 halves = 21760 B
#define CROW        66                        // padded f32 row stride (bank-conflict-free)

// ---------------------------------------------------------------------------
// Prep 1: combine base filters over nf (rotation is linear -> combine first).
// u[a][oc][ic][tap] = sum_n rw[n][a] * kernel[n][oc][ic][tap]
// ---------------------------------------------------------------------------
__global__ void combine_nf_kernel(const float* __restrict__ kern,
                                  const float* __restrict__ rw,
                                  float* __restrict__ u) {
    int idx = blockIdx.x * 256 + threadIdx.x;
    const int total = GROUP_ * OC_ * IC_ * TAPS;
    if (idx >= total) return;
    int t  = idx % TAPS;
    int ic = (idx / TAPS) % IC_;
    int oc = (idx / (TAPS * IC_)) % OC_;
    int a  = idx / (TAPS * IC_ * OC_);
    float s = 0.0f;
#pragma unroll
    for (int n = 0; n < NF_; ++n)
        s += rw[n * GROUP_ + a] * kern[(((n * OC_ + oc) * IC_ + ic) * TAPS) + t];
    u[idx] = s;
}

// ---------------------------------------------------------------------------
// Prep 2: bilinear-rotate by theta_a = -2*pi*a/8 (affine_grid/grid_sample,
// align_corners=False, zero pad) and pack f16 weights Wh[m][tap][ic], m=oc*8+a.
// ---------------------------------------------------------------------------
__global__ void rotate_pack_kernel(const float* __restrict__ u,
                                   _Float16* __restrict__ Wh) {
    int idx = blockIdx.x * 256 + threadIdx.x;
    const int total = CH_ * TAPS * IC_;
    if (idx >= total) return;
    int ic = idx % IC_;
    int t  = (idx / IC_) % TAPS;
    int m  = idx / (IC_ * TAPS);
    int a  = m % GROUP_;
    int oc = m / GROUP_;
    int ky = t / K_;
    int kx = t % K_;

    float theta = -2.0f * 3.14159265358979323846f * (float)a / (float)GROUP_;
    float c = cosf(theta), s = sinf(theta);
    float ysv = (2.0f * ky + 1.0f) / (float)K_ - 1.0f;  // pixel-center coords
    float xsv = (2.0f * kx + 1.0f) / (float)K_ - 1.0f;
    float gx = c * xsv - s * ysv;
    float gy = s * xsv + c * ysv;
    float ix = ((gx + 1.0f) * (float)K_ - 1.0f) * 0.5f;
    float iy = ((gy + 1.0f) * (float)K_ - 1.0f) * 0.5f;
    float x0f = floorf(ix), y0f = floorf(iy);
    float fx = ix - x0f, fy = iy - y0f;
    int xi = (int)x0f, yi = (int)y0f;

    const float* up = u + (size_t)((a * OC_ + oc) * IC_ + ic) * TAPS;
    auto g = [&](int yy, int xx) -> float {
        if (yy < 0 || yy >= K_ || xx < 0 || xx >= K_) return 0.0f;
        return up[yy * K_ + xx];
    };
    float v = g(yi,     xi    ) * (1.0f - fy) * (1.0f - fx)
            + g(yi,     xi + 1) * (1.0f - fy) * fx
            + g(yi + 1, xi    ) * fy          * (1.0f - fx)
            + g(yi + 1, xi + 1) * fy          * fx;
    Wh[idx] = (_Float16)v;
}

// ---------------------------------------------------------------------------
// Prep 3: NCHW fp32 -> padded channels-last f16: xh[b][y+2][x+2][ic].
// ---------------------------------------------------------------------------
__global__ void pack_input_kernel(const float* __restrict__ x,
                                  _Float16* __restrict__ xh) {
    size_t idx = (size_t)blockIdx.x * 256 + threadIdx.x;
    const size_t total = (size_t)B_ * PAD_HW * PAD_HW * IC_;
    if (idx >= total) return;
    int ic = (int)(idx % IC_);
    size_t p = idx / IC_;
    int xx = (int)(p % PAD_HW);
    int yy = (int)((p / PAD_HW) % PAD_HW);
    int b  = (int)(p / ((size_t)PAD_HW * PAD_HW));
    int sx = xx - 2, sy = yy - 2;
    float v = 0.0f;
    if (sx >= 0 && sx < HW_ && sy >= 0 && sy < HW_)
        v = x[(((size_t)b * IC_ + ic) * HW_ + sy) * HW_ + sx];
    xh[idx] = (_Float16)v;
}

// ---------------------------------------------------------------------------
// Main implicit-GEMM conv on v_wmma_f32_16x16x32_f16.
// Block = 128 threads = 4 waves; M-tile 64 (16/wave), N-tile 64 pixels.
// TDM stages the 5x68x32 f16 input patch into LDS once per block (one 2D
// descriptor, TENSORcnt-tracked), B fragments then come from LDS; the f32
// epilogue bounces through wave-private LDS for fully coalesced stores.
// ---------------------------------------------------------------------------
union FragH {
    uint4 u4[2];
    v16h  h;
};

__global__ __launch_bounds__(128)
void conv_wmma_kernel(const _Float16* __restrict__ Wh,
                      const _Float16* __restrict__ xh,
                      float* __restrict__ out) {
    __shared__ __align__(16) _Float16 patch[PATCH_E];      // 21760 B
    __shared__ __align__(16) float    cbuf[4][16 * CROW];  // 16896 B

    const int lane = threadIdx.x & 31;
    const int wave = threadIdx.x >> 5;
    const int hi   = lane >> 4;
    const int n    = lane & 15;

    const int b  = blockIdx.z;
    const int y  = blockIdx.x >> 1;            // output row 0..127
    const int x0 = (blockIdx.x & 1) * 64;      // pixel-tile start: 0 or 64
    const int m  = blockIdx.y * 64 + wave * 16 + n;

    // ---- TDM: one 2D tile load (5 rows x 2176 contiguous halves) ----------
    if (wave == 0) {
        const _Float16* gsrc =
            xh + ((size_t)b * PAD_HW + y) * PAD_HW * IC_ + (size_t)x0 * IC_;
        unsigned long long ga = (unsigned long long)(uintptr_t)gsrc;
        unsigned lds_off = (unsigned)(uintptr_t)(&patch[0]);

        v4u g0;
        g0[0] = 1u;                                   // count=1 (valid D#)
        g0[1] = lds_off;                              // lds_addr (bytes)
        g0[2] = (unsigned)(ga & 0xFFFFFFFFu);         // global_addr[31:0]
        g0[3] = (unsigned)(ga >> 32) | (2u << 30);    // global_addr[56:32] | type=2

        v8i g1;
        g1[0] = (int)(1u << 16);                      // data_size=1 -> 2-byte elems
        g1[1] = (int)((((unsigned)(PAD_HW * IC_)) & 0xFFFFu) << 16); // tensor_dim0 lo16 @ [63:48]
        g1[2] = (int)((((unsigned)(PAD_HW * IC_)) >> 16) |
                      ((((unsigned)PAD_HW) & 0xFFFFu) << 16));       // td0 hi16 | td1 lo16
        g1[3] = (int)(((((unsigned)PAD_HW) >> 16)) |
                      (((unsigned)PATCH_ROW_E) << 16));              // td1 hi16 | tile_dim0
        g1[4] = (int)K_;                               // tile_dim1=5, tile_dim2=0
        g1[5] = (int)(PAD_HW * IC_);                   // tensor_dim0_stride lo32
        g1[6] = 0;                                     // stride0 hi | stride1 lo
        g1[7] = 0;                                     // stride1 hi
        v4i gz4 = {0, 0, 0, 0};
        v8i gz8 = {0, 0, 0, 0, 0, 0, 0, 0};
        __builtin_amdgcn_tensor_load_to_lds(g0, g1, gz4, gz4, gz8, 0);
        __builtin_amdgcn_s_wait_tensorcnt(0);
    }
    __syncthreads();

    // ---- GEMM: 25 taps x K=32 input channels -------------------------------
    v8f acc0 = {}, acc1 = {}, acc2 = {}, acc3 = {};
    const _Float16* wrow = Wh + (size_t)m * (TAPS * IC_);

    for (int ky = 0; ky < K_; ++ky) {
        const _Float16* prow = patch + ky * PATCH_ROW_E;
#pragma unroll
        for (int kx = 0; kx < K_; ++kx) {
            const int t = ky * K_ + kx;
            FragH af;
            const _Float16* wp = wrow + t * IC_;
            af.u4[0] = *(const uint4*)(wp + hi * 16);
            af.u4[1] = *(const uint4*)(wp + hi * 16 + 8);

            const _Float16* xp = prow + (size_t)(kx + n) * IC_ + hi * 16;
            FragH bf;

            bf.u4[0] = *(const uint4*)(xp);
            bf.u4[1] = *(const uint4*)(xp + 8);
            acc0 = __builtin_amdgcn_wmma_f32_16x16x32_f16(false, af.h, false, bf.h,
                                                          (short)0, acc0, false, false);

            bf.u4[0] = *(const uint4*)(xp + 16 * IC_);
            bf.u4[1] = *(const uint4*)(xp + 16 * IC_ + 8);
            acc1 = __builtin_amdgcn_wmma_f32_16x16x32_f16(false, af.h, false, bf.h,
                                                          (short)0, acc1, false, false);

            bf.u4[0] = *(const uint4*)(xp + 32 * IC_);
            bf.u4[1] = *(const uint4*)(xp + 32 * IC_ + 8);
            acc2 = __builtin_amdgcn_wmma_f32_16x16x32_f16(false, af.h, false, bf.h,
                                                          (short)0, acc2, false, false);

            bf.u4[0] = *(const uint4*)(xp + 48 * IC_);
            bf.u4[1] = *(const uint4*)(xp + 48 * IC_ + 8);
            acc3 = __builtin_amdgcn_wmma_f32_16x16x32_f16(false, af.h, false, bf.h,
                                                          (short)0, acc3, false, false);
        }
    }

    // ---- Epilogue: leaky_relu, wave-private LDS transpose, coalesced store -
    float* cb = cbuf[wave];
    auto stash = [&](v8f acc, int ct) {
        int col = ct * 16 + n;
#pragma unroll
        for (int r = 0; r < 8; ++r) {
            float v = acc[r];
            v = v > 0.0f ? v : 0.01f * v;
            cb[(r + hi * 8) * CROW + col] = v;
        }
    };
    stash(acc0, 0);
    stash(acc1, 1);
    stash(acc2, 2);
    stash(acc3, 3);

    // same-wave DS ops are in-order; no barrier needed for wave-private buffer
    float* obase = out + (((size_t)b * CH_ + blockIdx.y * 64 + wave * 16) * HW_ + y) * HW_ + x0;
#pragma unroll
    for (int row = 0; row < 16; ++row) {
        float2 v = *(const float2*)&cb[row * CROW + lane * 2];
        *(float2*)&obase[(size_t)row * (HW_ * HW_) + lane * 2] = v;
    }
}

// ---------------------------------------------------------------------------
extern "C" void kernel_launch(void* const* d_in, const int* in_sizes, int n_in,
                              void* d_out, int out_size, void* d_ws, size_t ws_size,
                              hipStream_t stream) {
    const float* x    = (const float*)d_in[0];  // [16,32,128,128]
    const float* kern = (const float*)d_in[1];  // [8,64,32,5,5]
    const float* rw   = (const float*)d_in[2];  // [8,8]
    float* out = (float*)d_out;                 // [16,512,128,128] flat

    char* ws = (char*)d_ws;
    const size_t u_bytes  = (size_t)GROUP_ * OC_ * IC_ * TAPS * sizeof(float);   // 1,638,400
    const size_t wh_bytes = (size_t)CH_ * TAPS * IC_ * sizeof(_Float16);         //   819,200
    float*    u  = (float*)ws;
    _Float16* Wh = (_Float16*)(ws + u_bytes);
    _Float16* xh = (_Float16*)(ws + u_bytes + wh_bytes);  // 17,842,176 bytes

    {
        int total = GROUP_ * OC_ * IC_ * TAPS;             // 409,600
        combine_nf_kernel<<<(total + 255) / 256, 256, 0, stream>>>(kern, rw, u);
    }
    {
        int total = CH_ * TAPS * IC_;                      // 409,600
        rotate_pack_kernel<<<(total + 255) / 256, 256, 0, stream>>>(u, Wh);
    }
    {
        size_t total = (size_t)B_ * PAD_HW * PAD_HW * IC_; // 8,921,088
        pack_input_kernel<<<(unsigned)((total + 255) / 256), 256, 0, stream>>>(x, xh);
    }
    {
        dim3 grid(256, 8, 16);
        conv_wmma_kernel<<<grid, 128, 0, stream>>>(Wh, xh, out);
    }
}
